// EEGNeX_21268678050414
// MI455X (gfx1250) — matrix operand order repaired and checked by reading, not compiled, not run on hardware
//
#include <hip/hip_runtime.h>
#include <hip/hip_bf16.h>
#include <cstdint>

// ---------------------------------------------------------------------------
// EEGNeX forward for MI455X (gfx1250, wave32).
//  - LoRA folded into effective weights (exact: B@(A*x) == (B@A)*x).
//  - conv2+bn2+conv3+bn3+elu fused; conv2 runs as bf16 WMMA GEMM
//    (v_wmma_f32_16x16x32_bf16, f32 accumulate); the 524MB (B,F2,C,T)
//    tensor is never materialized.
//  - h1 intermediate stored as bf16 (65MB instead of 131MB).
//  - h1 slab + Weff2 staged via GLOBAL_LOAD_ASYNC_TO_LDS_B64 (ASYNCcnt),
//    double-buffered per-c so the DMA hides behind the 16-WMMA compute.
//  - two independent WMMA accumulators (even/odd K-chunks) to break the
//    D->C dependency chain.
// ---------------------------------------------------------------------------

#define BB    64
#define CC    64
#define TT    1000
#define NADP  109
#define RANK  4
#define F1    8
#define F2    32
#define F3    64
#define K12   64
#define K4    16
#define K5    16
#define NOUT  4
#define T3    250
#define T5    31
#define INF_  (F1 * T5)     // 248
#define EPS_  1e-5f
#define AOR   0.25f         // ALPHA / RANK

typedef __attribute__((ext_vector_type(16))) __bf16          v16bf;
typedef __attribute__((ext_vector_type(16))) unsigned short  v16us;
typedef __attribute__((ext_vector_type(8)))  float           v8f;

#if defined(__gfx1250__) && __has_builtin(__builtin_amdgcn_global_load_async_to_lds_b64)
#define USE_ASYNC_LDS 1
typedef int v2i_t __attribute__((vector_size(8)));
typedef __attribute__((address_space(1))) v2i_t glb_v2i;
typedef __attribute__((address_space(3))) v2i_t lds_v2i;
#if __has_builtin(__builtin_amdgcn_s_wait_asynccnt)
#define WAIT_ASYNC0() __builtin_amdgcn_s_wait_asynccnt(0)
#else
#define WAIT_ASYNC0() asm volatile("s_wait_asynccnt 0x0" ::: "memory")
#endif
#else
#define USE_ASYNC_LDS 0
#define WAIT_ASYNC0() ((void)0)
#endif

__device__ __forceinline__ unsigned short f2bf(float f) {
    unsigned int x = __float_as_uint(f);
    unsigned int r = x + 0x7FFFu + ((x >> 16) & 1u);   // RNE
    return (unsigned short)(r >> 16);
}
__device__ __forceinline__ float elu1(float v) {
    return v > 0.f ? v : (__expf(v) - 1.f);
}

// ---------------- workspace layout (bytes) ----------------
#define OFF_H1   ((size_t)0)                        // u16 : B*F1*C*T      = 32,768,000
#define OFF_W2E  (OFF_H1  + (size_t)65536000)       // u16 : B*F2*F1*K12  = 1,048,576
#define OFF_H3E  (OFF_W2E + (size_t)2097152)        // f32 : B*F3*T       = 4,096,000
#define OFF_H4   (OFF_H3E + (size_t)16384000)       // f32 : B*F3*T3      = 1,024,000
#define OFF_H5   (OFF_H4  + (size_t)4096000)        // f32 : B*F2*T3      =   512,000
#define OFF_H6   (OFF_H5  + (size_t)2048000)        // f32 : B*F1*T3      =   128,000
#define OFF_HP8  (OFF_H6  + (size_t)512000)         // f32 : B*F1*T5      =    15,872
#define OFF_W1E  (OFF_HP8 + (size_t)63488)          // f32 : B*F1*K12     =    32,768
#define OFF_L4E  (OFF_W1E + (size_t)131072)         // f32 : B*F2*F3*K4   = 2,097,152
#define OFF_L5E  (OFF_L4E + (size_t)8388608)        // f32 : B*F1*F2*K5   =   262,144
#define OFF_W3R  (OFF_L5E + (size_t)1048576)        // f32 : F3*C         =     4,096
#define OFF_WFR  (OFF_W3R + (size_t)16384)          // f32 : NOUT*INF     =       992

// ============================================================ renorm w3 (maxnorm 1.0)
__global__ void k_renorm_w3(const float* __restrict__ w3, float* __restrict__ w3r) {
    __shared__ float red[CC];
    __shared__ float sc;
    int f3 = blockIdx.x, c = threadIdx.x;
    float v = w3[f3 * CC + c];
    red[c] = v * v;
    __syncthreads();
    for (int s = CC / 2; s > 0; s >>= 1) {
        if (c < s) red[c] += red[c + s];
        __syncthreads();
    }
    if (c == 0) {
        float n = sqrtf(red[0]);
        sc = (n > 1.0f) ? (1.0f / (n + 1e-7f)) : 1.0f;
    }
    __syncthreads();
    w3r[f3 * CC + c] = v * sc;
}

// ============================================================ renorm wf (maxnorm 0.25)
__global__ void k_renorm_wf(const float* __restrict__ wf, float* __restrict__ wfr) {
    __shared__ float red[256];
    __shared__ float sc;
    int o = blockIdx.x, i = threadIdx.x;
    float v = (i < INF_) ? wf[o * INF_ + i] : 0.f;
    red[i] = v * v;
    __syncthreads();
    for (int s = 128; s > 0; s >>= 1) {
        if (i < s) red[i] += red[i + s];
        __syncthreads();
    }
    if (i == 0) {
        float n = sqrtf(red[0]);
        sc = (n > 0.25f) ? (0.25f / (n + 1e-7f)) : 1.0f;
    }
    __syncthreads();
    if (i < INF_) wfr[o * INF_ + i] = v * sc;
}

// ============================================================ effective weights
__global__ void k_weff1(const float* __restrict__ w1, const float* __restrict__ A1,
                        const float* __restrict__ B1, const int* __restrict__ sid,
                        float* __restrict__ W1E) {
    int b = blockIdx.x, i = threadIdx.x;               // 512 threads: f1*64+k
    int s = sid[b];
    int f1 = i >> 6, k = i & 63;
    float acc = w1[i];
#pragma unroll
    for (int r = 0; r < RANK; ++r)
        acc += AOR * B1[(s * F1 + f1) * RANK + r] * A1[(s * RANK + r) * K12 + k];
    W1E[b * (F1 * K12) + i] = acc;
}

__global__ void k_weff2(const float* __restrict__ w2, const float* __restrict__ A2,
                        const float* __restrict__ B2, const int* __restrict__ sid,
                        unsigned short* __restrict__ W2E) {
    int b = blockIdx.x, s = sid[b];
    for (int i = threadIdx.x; i < F2 * F1 * K12; i += blockDim.x) {
        int f2 = i >> 9, rem = i & 511, f1 = rem >> 6, k = rem & 63;
        float acc = w2[i];
#pragma unroll
        for (int r = 0; r < RANK; ++r)
            acc += AOR * B2[(s * F2 + f2) * RANK + r] *
                   A2[((s * RANK + r) * F1 + f1) * K12 + k];
        W2E[b * (F2 * F1 * K12) + i] = f2bf(acc);
    }
}

__global__ void k_leff4(const float* __restrict__ A4, const float* __restrict__ B4,
                        const int* __restrict__ sid, float* __restrict__ L4E) {
    int b = blockIdx.x, s = sid[b];
    for (int i = threadIdx.x; i < F2 * F3 * K4; i += blockDim.x) {
        int f2 = i >> 10, f3 = (i >> 4) & 63, k = i & 15;
        float acc = 0.f;
#pragma unroll
        for (int r = 0; r < RANK; ++r)
            acc += B4[(s * F2 + f2) * RANK + r] *
                   A4[((s * RANK + r) * F3 + f3) * K4 + k];
        L4E[b * (F2 * F3 * K4) + i] = AOR * acc;
    }
}

__global__ void k_leff5(const float* __restrict__ A5, const float* __restrict__ B5,
                        const int* __restrict__ sid, float* __restrict__ L5E) {
    int b = blockIdx.x, s = sid[b];
    for (int i = threadIdx.x; i < F1 * F2 * K5; i += blockDim.x) {
        int f1 = i >> 9, f2 = (i >> 4) & 31, k = i & 15;
        float acc = 0.f;
#pragma unroll
        for (int r = 0; r < RANK; ++r)
            acc += B5[(s * F1 + f1) * RANK + r] *
                   A5[((s * RANK + r) * F2 + f2) * K5 + k];
        L5E[b * (F1 * F2 * K5) + i] = AOR * acc;
    }
}

// ============================================================ conv1 (+folded LoRA) -> bf16 h1
__global__ void k_conv1(const float* __restrict__ x, const float* __restrict__ W1E,
                        unsigned short* __restrict__ H1) {
    __shared__ float sx[TT];
    __shared__ float sw[F1 * K12];
    int bc = blockIdx.x, b = bc >> 6, c = bc & 63;
    for (int i = threadIdx.x; i < TT; i += blockDim.x) sx[i] = x[(size_t)bc * TT + i];
    for (int i = threadIdx.x; i < F1 * K12; i += blockDim.x) sw[i] = W1E[b * (F1 * K12) + i];
    __syncthreads();
    for (int t = threadIdx.x; t < TT; t += blockDim.x) {
        float acc[F1] = {0, 0, 0, 0, 0, 0, 0, 0};
        for (int k = 0; k < K12; ++k) {
            int ti = t - 31 + k;
            float xv = (ti >= 0 && ti < TT) ? sx[ti] : 0.f;
#pragma unroll
            for (int f1 = 0; f1 < F1; ++f1) acc[f1] += xv * sw[f1 * K12 + k];
        }
#pragma unroll
        for (int f1 = 0; f1 < F1; ++f1)
            H1[(((size_t)b * F1 + f1) * CC + c) * TT + t] = f2bf(acc[f1]);
    }
}

// ============================================================ FUSED conv2(WMMA bf16) + bn2
//                                                              + conv3(w3r) + bn3 + elu
#define SLABW 132   // ushorts per f1 row (264B, 8B-multiple); indices 0..127 used
__global__ void k_conv2_wmma(const unsigned short* __restrict__ H1,
                             const unsigned short* __restrict__ W2E,
                             const float* __restrict__ w3r,
                             const float* __restrict__ g2, const float* __restrict__ b2,
                             const float* __restrict__ m2, const float* __restrict__ v2,
                             const float* __restrict__ g3, const float* __restrict__ b3,
                             const float* __restrict__ m3, const float* __restrict__ v3,
                             float* __restrict__ H3E) {
    __shared__ __attribute__((aligned(16))) unsigned short sW[F2 * F1 * K12];    // 32KB
    __shared__ __attribute__((aligned(16))) unsigned short sSlab[2][F1 * SLABW]; // 4.2KB
    __shared__ float sW3[F3 * CC];                    // 16KB  renormed w3
    __shared__ float sBnI[F2], sBnM[F2], sBnB[F2];

    const int tid  = threadIdx.x;
    const int lane = tid & 31;
    const int wv   = tid >> 5;          // 8 waves
    const int mT   = wv >> 2;           // 0..1  (f2 tile)
    const int nT   = wv & 3;            // 0..3  (t tile)
    const int half = lane >> 4;
    const int l16  = lane & 15;
    const int b    = blockIdx.y;
    const int t0   = blockIdx.x * 64;

    // ---- stage slab for channel c into buffer buf; slab[j] <-> tg = t0-32+j ----
    auto issue_slab = [&](int c, int buf) {
        const int f1 = tid >> 5;              // 8 rows x 32 lanes = 256 threads
        const int q  = (tid & 31) << 2;       // 4 ushorts (8B) per lane
        const int tg = t0 - 32 + q;
        const size_t gbase = (((size_t)b * F1 + f1) * CC + c) * (size_t)TT;
        unsigned short* dst = &sSlab[buf][f1 * SLABW + q];
        if (tg >= 0 && tg + 3 < TT) {
#if USE_ASYNC_LDS
            __builtin_amdgcn_global_load_async_to_lds_b64(
                (glb_v2i*)&H1[gbase + tg], (lds_v2i*)dst, 0, 0);
#else
#pragma unroll
            for (int u = 0; u < 4; ++u) dst[u] = H1[gbase + tg + u];
#endif
        } else {
#pragma unroll
            for (int u = 0; u < 4; ++u) {
                int tgu = tg + u;
                dst[u] = (tgu >= 0 && tgu < TT) ? H1[gbase + tgu] : (unsigned short)0;
            }
        }
    };

    // ---- one 16x16x32 WMMA step: A from sW, B from slab row ----
    auto mma = [&](int aoff, const unsigned short* row, int jb, v8f acc) -> v8f {
        union { v16us u; v16bf h; } Af, Bf;
#pragma unroll
        for (int v = 0; v < 8; ++v) {
            int kkA = ((v < 4) ? 0 : 16) + half * 8 + ((v & 3) << 1);
            Af.u[2 * v]     = sW[aoff + kkA];
            Af.u[2 * v + 1] = sW[aoff + kkA + 1];
            int j = jb + ((v < 4) ? 0 : 16) + half * 8 + ((v & 3) << 1);
            Bf.u[2 * v]     = row[j];
            Bf.u[2 * v + 1] = row[j + 1];
        }
        return __builtin_amdgcn_wmma_f32_16x16x32_bf16(
            false, Af.h, false, Bf.h, (short)0, acc, false, false);
    };

    // ---- preload: Weff2 (async), slab c=0, w3r, bn2 coeffs ----
#if USE_ASYNC_LDS
    for (int i = tid * 4; i < F2 * F1 * K12; i += 256 * 4)
        __builtin_amdgcn_global_load_async_to_lds_b64(
            (glb_v2i*)&W2E[b * (F2 * F1 * K12) + i], (lds_v2i*)&sW[i], 0, 0);
#else
    for (int i = tid; i < F2 * F1 * K12; i += 256) sW[i] = W2E[b * (F2 * F1 * K12) + i];
#endif
    issue_slab(0, 0);
    for (int i = tid; i < F3 * CC; i += 256) sW3[i] = w3r[i];
    if (tid < F2) {
        sBnI[tid] = g2[tid] / sqrtf(v2[tid] + EPS_);
        sBnM[tid] = m2[tid];
        sBnB[tid] = b2[tid];
    }
    WAIT_ASYNC0();
    __syncthreads();

    float accF3[16];
#pragma unroll
    for (int i = 0; i < 16; ++i) accF3[i] = 0.f;

    const int f2row = mT * 16 + l16;
    const int jb0   = 1 + nT * 16 + l16;     // +1: slab origin is t0-32, conv pad is 31

    for (int c = 0; c < CC; ++c) {
        const int cur = c & 1;
        if (c + 1 < CC) issue_slab(c + 1, cur ^ 1);   // DMA next slab during compute

        // conv2 tile: D(16x16) += Weff2(16x512) * im2col(512x16); K chunks of 32,
        // even/odd chunks on independent accumulators to break the WMMA chain.
        v8f accE = {0.f, 0.f, 0.f, 0.f, 0.f, 0.f, 0.f, 0.f};
        v8f accO = {0.f, 0.f, 0.f, 0.f, 0.f, 0.f, 0.f, 0.f};
#pragma unroll
        for (int ch = 0; ch < 16; ch += 2) {
            const unsigned short* row = &sSlab[cur][(ch >> 1) * SLABW];  // f1 = ch/2
            accE = mma(f2row * (F1 * K12) + ch * 32,       row, jb0,      accE);
            accO = mma(f2row * (F1 * K12) + (ch + 1) * 32, row, jb0 + 32, accO);
        }

        // bn2, then contract over c with depthwise w3 into F3 accumulators
#pragma unroll
        for (int j = 0; j < 8; ++j) {
            int f2 = mT * 16 + j + 8 * half;        // C-layout row
            float bnv = (accE[j] + accO[j] - sBnM[f2]) * sBnI[f2] + sBnB[f2];
            accF3[2 * j]     += sW3[(2 * f2) * CC + c]     * bnv;
            accF3[2 * j + 1] += sW3[(2 * f2 + 1) * CC + c] * bnv;
        }
        WAIT_ASYNC0();      // next slab landed
        __syncthreads();    // and everyone is done reading the current one
    }

    // bn3 + elu + store h3e (full-T resolution; pooled in next kernel)
    const int t = t0 + nT * 16 + l16;
    if (t < TT) {
#pragma unroll
        for (int j = 0; j < 8; ++j) {
            int f2 = mT * 16 + j + 8 * half;
#pragma unroll
            for (int s = 0; s < 2; ++s) {
                int f3 = 2 * f2 + s;
                float inv = g3[f3] / sqrtf(v3[f3] + EPS_);
                float val = (accF3[2 * j + s] - m3[f3]) * inv + b3[f3];
                H3E[((size_t)b * F3 + f3) * TT + t] = elu1(val);
            }
        }
    }
}

// ============================================================ avg-pool T: k=4, pad=1
__global__ void k_pool4(const float* __restrict__ H3E, float* __restrict__ H4) {
    int row = blockIdx.x;                 // b*F3 + f3
    int t3 = threadIdx.x;
    if (t3 >= T3) return;
    float s = 0.f;
#pragma unroll
    for (int u = 0; u < 4; ++u) {
        int ti = 4 * t3 - 1 + u;
        if (ti >= 0 && ti < TT) s += H3E[(size_t)row * TT + ti];
    }
    H4[(size_t)row * T3 + t3] = s * 0.25f;
}

// ============================================================ conv4 (dil2) + LoRA(dil1) + bn4
__global__ void k_conv4(const float* __restrict__ H4, const float* __restrict__ w4,
                        const float* __restrict__ L4E,
                        const float* __restrict__ g4, const float* __restrict__ b4,
                        const float* __restrict__ m4, const float* __restrict__ v4,
                        float* __restrict__ H5) {
    __shared__ float sin[F3 * T3];        // 62.5KB
    int b = blockIdx.x;
    for (int i = threadIdx.x; i < F3 * T3; i += blockDim.x)
        sin[i] = H4[(size_t)b * F3 * T3 + i];
    __syncthreads();
    int t = threadIdx.x;
    if (t >= T3) return;
    for (int f2 = 0; f2 < F2; ++f2) {
        float acc = 0.f;
#pragma unroll 1
        for (int f3 = 0; f3 < F3; ++f3) {
            const float* wr = w4 + (f2 * F3 + f3) * K4;
            const float* lr = L4E + (((size_t)b * F2 + f2) * F3 + f3) * K4;
            const float* ir = sin + f3 * T3;
#pragma unroll
            for (int k = 0; k < K4; ++k) {
                int ti = t - 15 + 2 * k;                 // dil 2, pad (15,15)
                if (ti >= 0 && ti < T3) acc += ir[ti] * wr[k];
                int tj = t - 7 + k;                      // dil 1, pad (7,8)
                if (tj >= 0 && tj < T3) acc += ir[tj] * lr[k];
            }
        }
        float inv = g4[f2] / sqrtf(v4[f2] + EPS_);
        H5[((size_t)b * F2 + f2) * T3 + t] = (acc - m4[f2]) * inv + b4[f2];
    }
}

// ============================================================ conv5 (dil4) + LoRA(dil1) + bn5 + elu
__global__ void k_conv5(const float* __restrict__ H5, const float* __restrict__ w5,
                        const float* __restrict__ L5E,
                        const float* __restrict__ g5, const float* __restrict__ b5,
                        const float* __restrict__ m5, const float* __restrict__ v5,
                        float* __restrict__ H6) {
    __shared__ float sin[F2 * T3];        // 31.25KB
    int b = blockIdx.x;
    for (int i = threadIdx.x; i < F2 * T3; i += blockDim.x)
        sin[i] = H5[(size_t)b * F2 * T3 + i];
    __syncthreads();
    int t = threadIdx.x;
    if (t >= T3) return;
    for (int f1 = 0; f1 < F1; ++f1) {
        float acc = 0.f;
#pragma unroll 1
        for (int f2 = 0; f2 < F2; ++f2) {
            const float* wr = w5 + (f1 * F2 + f2) * K5;
            const float* lr = L5E + (((size_t)b * F1 + f1) * F2 + f2) * K5;
            const float* ir = sin + f2 * T3;
#pragma unroll
            for (int k = 0; k < K5; ++k) {
                int ti = t - 30 + 4 * k;                 // dil 4, pad (30,30)
                if (ti >= 0 && ti < T3) acc += ir[ti] * wr[k];
                int tj = t - 7 + k;                      // dil 1, pad (7,8)
                if (tj >= 0 && tj < T3) acc += ir[tj] * lr[k];
            }
        }
        float inv = g5[f1] / sqrtf(v5[f1] + EPS_);
        H6[((size_t)b * F1 + f1) * T3 + t] = elu1((acc - m5[f1]) * inv + b5[f1]);
    }
}

// ============================================================ avg-pool T: k=8, pad=1
__global__ void k_pool8(const float* __restrict__ H6, float* __restrict__ HP8) {
    int row = blockIdx.x;                 // b*F1 + f1
    int t5 = threadIdx.x;
    if (t5 >= T5) return;
    float s = 0.f;
#pragma unroll
    for (int u = 0; u < 8; ++u) {
        int ti = 8 * t5 - 1 + u;
        if (ti >= 0 && ti < T3) s += H6[(size_t)row * T3 + ti];
    }
    HP8[(size_t)row * T5 + t5] = s * 0.125f;
}

// ============================================================ final linear with renormed wf
__global__ void k_final(const float* __restrict__ HP8, const float* __restrict__ wfr,
                        const float* __restrict__ bf, float* __restrict__ out) {
    __shared__ float red[256];
    int b = blockIdx.x;
    int o = threadIdx.x >> 6, i0 = threadIdx.x & 63;
    float p = 0.f;
    for (int i = i0; i < INF_; i += 64)
        p += HP8[(size_t)b * INF_ + i] * wfr[o * INF_ + i];
    red[threadIdx.x] = p;
    __syncthreads();
    for (int s = 32; s > 0; s >>= 1) {
        if (i0 < s) red[threadIdx.x] += red[threadIdx.x + s];
        __syncthreads();
    }
    if (i0 == 0) out[b * NOUT + o] = red[threadIdx.x] + bf[o];
}

// ============================================================ launcher
extern "C" void kernel_launch(void* const* d_in, const int* in_sizes, int n_in,
                              void* d_out, int out_size, void* d_ws, size_t ws_size,
                              hipStream_t stream) {
    (void)in_sizes; (void)n_in; (void)out_size; (void)ws_size;
    const float* x   = (const float*)d_in[0];
    const int*   sid = (const int*)  d_in[1];
    const float* w1  = (const float*)d_in[2];
    const float* A1  = (const float*)d_in[3];
    const float* B1  = (const float*)d_in[4];
    const float* w2  = (const float*)d_in[5];
    const float* A2  = (const float*)d_in[6];
    const float* B2  = (const float*)d_in[7];
    const float* w3  = (const float*)d_in[8];
    const float* w4  = (const float*)d_in[9];
    const float* A4  = (const float*)d_in[10];
    const float* B4  = (const float*)d_in[11];
    const float* w5  = (const float*)d_in[12];
    const float* A5  = (const float*)d_in[13];
    const float* B5  = (const float*)d_in[14];
    const float* g2 = (const float*)d_in[15], *b2 = (const float*)d_in[16];
    const float* m2 = (const float*)d_in[17], *v2 = (const float*)d_in[18];
    const float* g3 = (const float*)d_in[19], *b3 = (const float*)d_in[20];
    const float* m3 = (const float*)d_in[21], *v3 = (const float*)d_in[22];
    const float* g4 = (const float*)d_in[23], *b4 = (const float*)d_in[24];
    const float* m4 = (const float*)d_in[25], *v4 = (const float*)d_in[26];
    const float* g5 = (const float*)d_in[27], *b5 = (const float*)d_in[28];
    const float* m5 = (const float*)d_in[29], *v5 = (const float*)d_in[30];
    const float* wf = (const float*)d_in[31];
    const float* bf = (const float*)d_in[32];
    float* out = (float*)d_out;

    char* ws = (char*)d_ws;
    unsigned short* H1  = (unsigned short*)(ws + OFF_H1);
    unsigned short* W2E = (unsigned short*)(ws + OFF_W2E);
    float* H3E = (float*)(ws + OFF_H3E);
    float* H4  = (float*)(ws + OFF_H4);
    float* H5  = (float*)(ws + OFF_H5);
    float* H6  = (float*)(ws + OFF_H6);
    float* HP8 = (float*)(ws + OFF_HP8);
    float* W1E = (float*)(ws + OFF_W1E);
    float* L4E = (float*)(ws + OFF_L4E);
    float* L5E = (float*)(ws + OFF_L5E);
    float* W3R = (float*)(ws + OFF_W3R);
    float* WFR = (float*)(ws + OFF_WFR);

    k_renorm_w3<<<F3, CC, 0, stream>>>(w3, W3R);
    k_renorm_wf<<<NOUT, 256, 0, stream>>>(wf, WFR);
    k_weff1<<<BB, F1 * K12, 0, stream>>>(w1, A1, B1, sid, W1E);
    k_weff2<<<BB, 256, 0, stream>>>(w2, A2, B2, sid, W2E);
    k_leff4<<<BB, 256, 0, stream>>>(A4, B4, sid, L4E);
    k_leff5<<<BB, 256, 0, stream>>>(A5, B5, sid, L5E);

    k_conv1<<<BB * CC, 256, 0, stream>>>(x, W1E, H1);

    dim3 g2d(16, BB);   // 16 t-tiles of 64, per sample
    k_conv2_wmma<<<g2d, 256, 0, stream>>>(H1, W2E, W3R,
                                          g2, b2, m2, v2, g3, b3, m3, v3, H3E);

    k_pool4<<<BB * F3, 256, 0, stream>>>(H3E, H4);
    k_conv4<<<BB, 256, 0, stream>>>(H4, w4, L4E, g4, b4, m4, v4, H5);
    k_conv5<<<BB, 256, 0, stream>>>(H5, w5, L5E, g5, b5, m5, v5, H6);
    k_pool8<<<BB * F1, 32, 0, stream>>>(H6, HP8);
    k_final<<<BB, 256, 0, stream>>>(HP8, WFR, bf, out);
}
